// S4Model_19550691131550
// MI455X (gfx1250) — compile-verified, matrix operand order checked by reading, not run on hardware
//
#include <hip/hip_runtime.h>
#include <math.h>

// Model dims
#define B_    32
#define L_    2048
#define DIN_  128
#define H_    512
#define N_    32
#define NL_   6
#define DOUT_ 10

// Scan chunking
#define CH_   8
#define LC_   (L_ / CH_)   // 256

typedef __attribute__((ext_vector_type(16))) __bf16 v16bf;
typedef __attribute__((ext_vector_type(8)))  __bf16 v8bf;
typedef __attribute__((ext_vector_type(8)))  float  v8f;

// ---------------------------------------------------------------------------
// CDNA5 async global->LDS copy (per-lane 16B), tracked by ASYNCcnt.
// ---------------------------------------------------------------------------
__device__ inline void async_load_b128(unsigned ldsAddr, const void* gaddr) {
    asm volatile("global_load_async_to_lds_b128 %0, %1, off"
                 :: "v"(ldsAddr), "v"((unsigned long long)(size_t)gaddr)
                 : "memory");
}
__device__ inline void wait_asynccnt0() {
    asm volatile("s_wait_asynccnt 0" ::: "memory");
}

__device__ inline float gelu_exact(float v) {
    return 0.5f * v * (1.0f + erff(v * 0.70710678118f));
}

// ---------------------------------------------------------------------------
// Precompute per-(layer,h,n) scan constants:
//   d   = exp(dt*A)                  (complex per-step decay)
//   dP  = exp(LC*dt*A) = d^LC        (per-chunk decay, for state handoff)
//   c2  = 2 * C * (exp(dt*A)-1)/A
// ---------------------------------------------------------------------------
__global__ void precompute_consts(const float* __restrict__ log_dt,
                                  const float* __restrict__ log_A_real,
                                  const float* __restrict__ A_imag,
                                  const float* __restrict__ C_re,
                                  const float* __restrict__ C_im,
                                  float* __restrict__ d_re, float* __restrict__ d_im,
                                  float* __restrict__ dp_re, float* __restrict__ dp_im,
                                  float* __restrict__ c2_re, float* __restrict__ c2_im) {
    int idx = blockIdx.x * blockDim.x + threadIdx.x;      // over NL*H*N
    if (idx >= NL_ * H_ * N_) return;
    int hn = idx % (H_ * N_);
    int i  = idx / (H_ * N_);
    int h  = hn / N_;
    float dt  = expf(log_dt[i * H_ + h]);
    float Are = -expf(log_A_real[idx]);
    float Aim = A_imag[idx];
    float xre = dt * Are, xim = dt * Aim;
    float e   = expf(xre);
    float dre = e * cosf(xim);
    float dim = e * sinf(xim);
    float ep  = expf((float)LC_ * xre);
    dp_re[idx] = ep * cosf((float)LC_ * xim);
    dp_im[idx] = ep * sinf((float)LC_ * xim);
    // (exp(dtA)-1)/A  =  num * conj(A) / |A|^2
    float nre = dre - 1.0f, nim = dim;
    float inv = 1.0f / (Are * Are + Aim * Aim);
    float qre = (nre * Are + nim * Aim) * inv;
    float qim = (nim * Are - nre * Aim) * inv;
    float cr = C_re[idx], ci = C_im[idx];
    c2_re[idx] = 2.0f * (cr * qre - ci * qim);
    c2_im[idx] = 2.0f * (cr * qim + ci * qre);
    d_re[idx]  = dre;
    d_im[idx]  = dim;
}

// ---------------------------------------------------------------------------
// One-shot fp32 -> bf16 weight conversion (all layers of out_W).
// ---------------------------------------------------------------------------
__global__ __launch_bounds__(256) void convert_w(const float* __restrict__ src,
                                                 __bf16* __restrict__ dst, int n) {
    int idx = blockIdx.x * blockDim.x + threadIdx.x;
    if (idx < n) dst[idx] = (__bf16)src[idx];
}

// ---------------------------------------------------------------------------
// WMMA 16x32 bf16 operand gather from fp32 global (gfx1250 wave32 layout):
//  lanes 0-15:  row = lane,    K-half base 0
//  lanes 16-31: row = lane-16, K-half base 8
//  vector elems 0..7 -> K = kb+0..7 ; elems 8..15 -> K = 16+kb+0..7
// ---------------------------------------------------------------------------
__device__ inline v16bf load_tile16x32_f32(const float* __restrict__ src, int ld,
                                           int r0, int k0, int lane) {
    int r  = r0 + (lane & 15);
    int kb = (lane >> 4) << 3;
    const float* p = src + (size_t)r * ld + k0 + kb;
    v16bf t;
#pragma unroll
    for (int j = 0; j < 8; ++j) {
        t[j]     = (__bf16)p[j];
        t[8 + j] = (__bf16)p[16 + j];
    }
    return t;
}

// Same gather from bf16 memory (global or LDS): two aligned 16B vector loads.
__device__ inline v16bf load_tile16x32_bf16(const __bf16* __restrict__ src, int ld,
                                            int r0, int k0, int lane) {
    int r  = r0 + (lane & 15);
    int kb = (lane >> 4) << 3;
    const __bf16* p = src + (size_t)r * ld + k0 + kb;
    v8bf lo = *(const v8bf*)(p);
    v8bf hi = *(const v8bf*)(p + 16);
    return __builtin_shufflevector(lo, hi, 0, 1, 2, 3, 4, 5, 6, 7,
                                   8, 9, 10, 11, 12, 13, 14, 15);
}

// ---------------------------------------------------------------------------
// Encoder GEMM: u[(b,l), h] = x[(b,l), :] . enc_W[h, :] + enc_b[h]
// M = B*L, N = H, K = DIN. Each wave: 16x64 output strip (4 accumulators).
// ---------------------------------------------------------------------------
__global__ __launch_bounds__(256) void gemm_encode(const float* __restrict__ x,
                                                   const float* __restrict__ W,
                                                   const float* __restrict__ bias,
                                                   float* __restrict__ u) {
    int wave = (blockIdx.x * blockDim.x + threadIdx.x) >> 5;
    int lane = threadIdx.x & 31;
    const int nStrips = H_ / 64;                // 8
    int mt = wave / nStrips;
    int ns = wave % nStrips;

    v8f acc[4] = {};
#pragma unroll
    for (int k0 = 0; k0 < DIN_; k0 += 32) {
        v16bf a = load_tile16x32_f32(x, DIN_, mt * 16, k0, lane);
#pragma unroll
        for (int t = 0; t < 4; ++t) {
            v16bf b = load_tile16x32_f32(W, DIN_, (ns * 4 + t) * 16, k0, lane);
            acc[t] = __builtin_amdgcn_wmma_f32_16x16x32_bf16(false, a, false, b,
                                                             (short)0, acc[t], false, false);
        }
    }
    int nc  = lane & 15;
    int row = mt * 16 + ((lane >> 4) << 3);
#pragma unroll
    for (int t = 0; t < 4; ++t) {
        int n = ns * 64 + t * 16 + nc;
        float bv = bias[n];
#pragma unroll
        for (int r = 0; r < 8; ++r)
            u[(size_t)(row + r) * H_ + n] = acc[t][r] + bv;
    }
}

// ---------------------------------------------------------------------------
// Pointwise conv GEMM, all-bf16 operands:
// y2[(b,l), o] = sum_h Wb[o,h] * yb[(b,l), h] + out_b[o]
// yb already holds gelu(conv + D*u) in bf16; Wb is bf16 out_W.
// Block tile 128x64 (8 waves x 16 rows). Weight tile double-buffered in LDS
// via CDNA5 async global->LDS copies; one barrier per k-step.
// ---------------------------------------------------------------------------
__global__ __launch_bounds__(256) void gemm_out(const __bf16* __restrict__ yb,
                                                const __bf16* __restrict__ Wball,
                                                const float* __restrict__ ball,
                                                float* __restrict__ y2,
                                                int layer) {
    __shared__ __attribute__((aligned(16))) __bf16 Bs[2][64 * 32];  // [n][k], 4KB each
    int lane = threadIdx.x & 31;
    int wave = threadIdx.x >> 5;                // m-tile within block (0..7)
    int mb   = blockIdx.x;                      // 0..(B*L/128 - 1)
    int nb   = blockIdx.y;                      // 0..(2H/64 - 1)
    const __bf16* Wl = Wball + (size_t)layer * 2 * H_ * H_;
    const float*  bl = ball  + (size_t)layer * 2 * H_;
    unsigned base[2] = { (unsigned)(size_t)&Bs[0][0], (unsigned)(size_t)&Bs[1][0] };

    // Each thread stages one 16B unit of the 64x32 bf16 tile (4KB total).
    int tn  = threadIdx.x >> 2;                 // 0..63  (row n)
    int tkk = (threadIdx.x & 3) << 3;           // 0,8,16,24 (bf16 elems)
    const __bf16* wrow = Wl + (size_t)(nb * 64 + tn) * H_ + tkk;
    unsigned ldst = (unsigned)((tn * 32 + tkk) * 2);

    // Prologue: stage k0 = 0 into buffer 0.
    async_load_b128(base[0] + ldst, wrow);

    v8f acc[4] = {};
    int cur = 0;
    for (int k0 = 0; k0 < H_; k0 += 32) {
        wait_asynccnt0();
        __syncthreads();                        // tile ready; prev buffer free
        if (k0 + 32 < H_)
            async_load_b128(base[cur ^ 1] + ldst, wrow + k0 + 32);

        v16bf a = load_tile16x32_bf16(yb, H_, (mb * 8 + wave) * 16, k0, lane);
        const __bf16* bs = &Bs[cur][0];
#pragma unroll
        for (int t = 0; t < 4; ++t) {
            v16bf b = load_tile16x32_bf16(bs, 32, t * 16, 0, lane);
            acc[t] = __builtin_amdgcn_wmma_f32_16x16x32_bf16(false, a, false, b,
                                                             (short)0, acc[t], false, false);
        }
        cur ^= 1;
    }
    int nc  = lane & 15;
    int row = (mb * 8 + wave) * 16 + ((lane >> 4) << 3);
#pragma unroll
    for (int t = 0; t < 4; ++t) {
        int n = nb * 64 + t * 16 + nc;
        float bv = bl[n];
#pragma unroll
        for (int r = 0; r < 8; ++r)
            y2[(size_t)(row + r) * (2 * H_) + n] = acc[t][r] + bv;
    }
}

// ---------------------------------------------------------------------------
// Chunked S4D scan, phase A: local final state of each L-chunk (zero init).
// One wave per (b, h-block, chunk). lane = h.
// ---------------------------------------------------------------------------
__global__ __launch_bounds__(32, 1) void scan_states(const float* __restrict__ u,
                                                     const float* __restrict__ d_re,
                                                     const float* __restrict__ d_im,
                                                     float* __restrict__ Sre,
                                                     float* __restrict__ Sim,
                                                     int layer) {
    int lane = threadIdx.x;
    int h    = blockIdx.x * 32 + lane;
    int b    = blockIdx.y;
    int j    = blockIdx.z;
    size_t cbase = ((size_t)layer * H_ + h) * N_;

    float dre[N_], dim_[N_], sre[N_], sim[N_];
#pragma unroll
    for (int n = 0; n < N_; ++n) {
        dre[n]  = d_re[cbase + n];
        dim_[n] = d_im[cbase + n];
        sre[n]  = 0.0f;
        sim[n]  = 0.0f;
    }
    const float* up = u + ((size_t)b * L_ + (size_t)j * LC_) * H_ + h;
    for (int l = 0; l < LC_; ++l) {
        float uv = up[(size_t)l * H_];
#pragma unroll
        for (int n = 0; n < N_; ++n) {
            float nr = dre[n] * sre[n] - dim_[n] * sim[n] + uv;
            float ni = dre[n] * sim[n] + dim_[n] * sre[n];
            sre[n] = nr;
            sim[n] = ni;
        }
    }
    size_t sb = (((size_t)b * CH_ + j) * H_ + h) * N_;
#pragma unroll
    for (int n = 0; n < N_; ++n) {
        Sre[sb + n] = sre[n];
        Sim[sb + n] = sim[n];
    }
}

// ---------------------------------------------------------------------------
// Phase B: tiny sequential prefix across the CH chunks per (b,h,n).
// In-place: local-final states are replaced by chunk *start* states.
//   Sstart_0 = 0 ; Sstart_j = d^LC * Sstart_{j-1} + Slocal_{j-1}
// ---------------------------------------------------------------------------
__global__ __launch_bounds__(256) void scan_prefix(const float* __restrict__ dp_re,
                                                   const float* __restrict__ dp_im,
                                                   float* __restrict__ Sre,
                                                   float* __restrict__ Sim,
                                                   int layer) {
    int idx = blockIdx.x * blockDim.x + threadIdx.x;  // B*H*N
    if (idx >= B_ * H_ * N_) return;
    int n = idx % N_;
    int h = (idx / N_) % H_;
    int b = idx / (N_ * H_);
    size_t cidx = ((size_t)layer * H_ + h) * N_ + n;
    float pr = dp_re[cidx], pi = dp_im[cidx];
    float ar = 0.0f, ai = 0.0f;                       // Sstart_0
    for (int j = 0; j < CH_; ++j) {
        size_t s = (((size_t)b * CH_ + j) * H_ + h) * N_ + n;
        float lr = Sre[s], li = Sim[s];               // local final of chunk j
        Sre[s] = ar;                                  // overwrite with start
        Sim[s] = ai;
        float nr = pr * ar - pi * ai + lr;
        float ni = pr * ai + pi * ar + li;
        ar = nr;
        ai = ni;
    }
}

// ---------------------------------------------------------------------------
// Phase C: full scan of each chunk seeded with its true start state, with
// fused skip connection + exact GELU; emits bf16 A-operand for the GEMM.
//   yb[b,l,h] = bf16( gelu( D[h]*u + Re(sum_n c2_n * s_n) ) )
// ---------------------------------------------------------------------------
__global__ __launch_bounds__(32, 1) void scan_output(const float* __restrict__ u,
                                                     __bf16* __restrict__ yb,
                                                     const float* __restrict__ d_re,
                                                     const float* __restrict__ d_im,
                                                     const float* __restrict__ c2_re,
                                                     const float* __restrict__ c2_im,
                                                     const float* __restrict__ Sre,
                                                     const float* __restrict__ Sim,
                                                     const float* __restrict__ Dskip,
                                                     int layer) {
    int lane = threadIdx.x;
    int h    = blockIdx.x * 32 + lane;
    int b    = blockIdx.y;
    int j    = blockIdx.z;
    size_t cbase = ((size_t)layer * H_ + h) * N_;
    size_t sb    = (((size_t)b * CH_ + j) * H_ + h) * N_;

    float dre[N_], dim_[N_], cre[N_], cim[N_], sre[N_], sim[N_];
#pragma unroll
    for (int n = 0; n < N_; ++n) {
        dre[n]  = d_re[cbase + n];
        dim_[n] = d_im[cbase + n];
        cre[n]  = c2_re[cbase + n];
        cim[n]  = c2_im[cbase + n];
        sre[n]  = Sre[sb + n];
        sim[n]  = Sim[sb + n];
    }
    float dsk = Dskip[layer * H_ + h];

    const float* up = u  + ((size_t)b * L_ + (size_t)j * LC_) * H_ + h;
    __bf16*      yp = yb + ((size_t)b * L_ + (size_t)j * LC_) * H_ + h;
    for (int l = 0; l < LC_; ++l) {
        float uv  = up[(size_t)l * H_];
        float acc = dsk * uv;
#pragma unroll
        for (int n = 0; n < N_; ++n) {
            float nr = dre[n] * sre[n] - dim_[n] * sim[n] + uv;
            float ni = dre[n] * sim[n] + dim_[n] * sre[n];
            sre[n] = nr;
            sim[n] = ni;
            acc += cre[n] * nr - cim[n] * ni;
        }
        yp[(size_t)l * H_] = (__bf16)gelu_exact(acc);
    }
}

// ---------------------------------------------------------------------------
// Fused GLU + residual + LayerNorm(H). One wave per (b,l) row; 16 elems/lane.
// ---------------------------------------------------------------------------
__global__ __launch_bounds__(256) void glu_res_ln(const float* __restrict__ y2,
                                                  float* __restrict__ u,
                                                  const float* __restrict__ ln_w,
                                                  const float* __restrict__ ln_b,
                                                  int layer) {
    int lane = threadIdx.x & 31;
    int wave = threadIdx.x >> 5;
    size_t row = (size_t)blockIdx.x * 8 + wave;       // (b*L + l)
    const float* a = y2 + row * (2 * H_);
    const float* g = a + H_;
    float*       up = u + row * H_;

    float v[H_ / 32];
    float sum = 0.0f;
#pragma unroll
    for (int j = 0; j < H_ / 32; ++j) {
        int h = lane + j * 32;
        float av = a[h], gv = g[h];
        float z  = av * (1.0f / (1.0f + expf(-gv)));
        float t  = z + up[h];
        v[j] = t;
        sum += t;
    }
#pragma unroll
    for (int m = 16; m >= 1; m >>= 1) sum += __shfl_xor(sum, m, 32);
    float mu  = sum * (1.0f / H_);
    float var = 0.0f;
#pragma unroll
    for (int j = 0; j < H_ / 32; ++j) {
        float d = v[j] - mu;
        var += d * d;
    }
#pragma unroll
    for (int m = 16; m >= 1; m >>= 1) var += __shfl_xor(var, m, 32);
    float inv = rsqrtf(var * (1.0f / H_) + 1e-5f);
    const float* w  = ln_w + layer * H_;
    const float* bb = ln_b + layer * H_;
#pragma unroll
    for (int j = 0; j < H_ / 32; ++j) {
        int h = lane + j * 32;
        up[h] = (v[j] - mu) * inv * w[h] + bb[h];
    }
}

// ---------------------------------------------------------------------------
// Mean pool over L, then tiny decoder GEMV.
// ---------------------------------------------------------------------------
__global__ __launch_bounds__(256) void mean_pool(const float* __restrict__ u,
                                                 float* __restrict__ pooled) {
    int idx = blockIdx.x * blockDim.x + threadIdx.x;  // B*H
    int b = idx / H_, h = idx % H_;
    const float* up = u + (size_t)b * L_ * H_ + h;
    float s = 0.0f;
    for (int l = 0; l < L_; ++l) s += up[(size_t)l * H_];
    pooled[idx] = s * (1.0f / L_);
}

__global__ __launch_bounds__(256) void decode(const float* __restrict__ pooled,
                                              const float* __restrict__ W,
                                              const float* __restrict__ bias,
                                              float* __restrict__ out) {
    int idx = blockIdx.x * blockDim.x + threadIdx.x;
    if (idx >= B_ * DOUT_) return;
    int b = idx / DOUT_, o = idx % DOUT_;
    float s = bias[o];
    for (int h = 0; h < H_; ++h) s += pooled[b * H_ + h] * W[o * H_ + h];
    out[idx] = s;
}

// ---------------------------------------------------------------------------
extern "C" void kernel_launch(void* const* d_in, const int* in_sizes, int n_in,
                              void* d_out, int out_size, void* d_ws, size_t ws_size,
                              hipStream_t stream) {
    const float* x          = (const float*)d_in[0];
    const float* enc_W      = (const float*)d_in[1];
    const float* enc_b      = (const float*)d_in[2];
    const float* log_dt     = (const float*)d_in[3];
    const float* log_A_real = (const float*)d_in[4];
    const float* A_imag     = (const float*)d_in[5];
    const float* C_re       = (const float*)d_in[6];
    const float* C_im       = (const float*)d_in[7];
    const float* Dskip      = (const float*)d_in[8];
    const float* out_W      = (const float*)d_in[9];
    const float* out_b      = (const float*)d_in[10];
    const float* ln_w       = (const float*)d_in[11];
    const float* ln_b       = (const float*)d_in[12];
    const float* dec_W      = (const float*)d_in[13];
    const float* dec_b      = (const float*)d_in[14];
    float* out = (float*)d_out;

    // Workspace carve-up (fp32 words; every block is a multiple of 4 words,
    // so all regions stay 16B-aligned for b128 access).
    float* ws = (float*)d_ws;
    size_t off = 0;
    float* u      = ws + off; off += (size_t)B_ * L_ * H_;
    float* y2     = ws + off; off += (size_t)B_ * L_ * 2 * H_;
    float* dre    = ws + off; off += (size_t)NL_ * H_ * N_;
    float* dim    = ws + off; off += (size_t)NL_ * H_ * N_;
    float* dpre   = ws + off; off += (size_t)NL_ * H_ * N_;
    float* dpim   = ws + off; off += (size_t)NL_ * H_ * N_;
    float* c2re   = ws + off; off += (size_t)NL_ * H_ * N_;
    float* c2im   = ws + off; off += (size_t)NL_ * H_ * N_;
    float* Sre    = ws + off; off += (size_t)B_ * CH_ * H_ * N_;
    float* Sim    = ws + off; off += (size_t)B_ * CH_ * H_ * N_;
    float* pooled = ws + off; off += (size_t)B_ * H_;
    __bf16* Wb    = (__bf16*)(ws + off); off += (size_t)NL_ * 2 * H_ * H_ / 2;
    __bf16* yb    = (__bf16*)(ws + off); off += (size_t)B_ * L_ * H_ / 2;

    // 1) scan constants + bf16 weights
    {
        int total = NL_ * H_ * N_;
        precompute_consts<<<(total + 255) / 256, 256, 0, stream>>>(
            log_dt, log_A_real, A_imag, C_re, C_im,
            dre, dim, dpre, dpim, c2re, c2im);
        int wtotal = NL_ * 2 * H_ * H_;
        convert_w<<<(wtotal + 255) / 256, 256, 0, stream>>>(out_W, Wb, wtotal);
    }
    // 2) encoder GEMM (bf16 WMMA, f32 accum)
    {
        int waves = ((B_ * L_) / 16) * (H_ / 64);   // 32768
        gemm_encode<<<waves / 8, 256, 0, stream>>>(x, enc_W, enc_b, u);
    }
    // 3) layers
    for (int i = 0; i < NL_; ++i) {
        dim3 sgrid(H_ / 32, B_, CH_);
        scan_states<<<sgrid, 32, 0, stream>>>(u, dre, dim, Sre, Sim, i);
        scan_prefix<<<(B_ * H_ * N_) / 256, 256, 0, stream>>>(dpre, dpim, Sre, Sim, i);
        scan_output<<<sgrid, 32, 0, stream>>>(u, yb, dre, dim, c2re, c2im,
                                              Sre, Sim, Dskip, i);

        dim3 ggrid((B_ * L_) / 128, (2 * H_) / 64);  // 512 x 16
        gemm_out<<<ggrid, 256, 0, stream>>>(yb, Wb, out_b, y2, i);

        glu_res_ln<<<(B_ * L_) / 8, 256, 0, stream>>>(y2, u, ln_w, ln_b, i);
    }
    // 4) pool + decode
    mean_pool<<<(B_ * H_) / 256, 256, 0, stream>>>(u, pooled);
    decode<<<(B_ * DOUT_ + 255) / 256, 256, 0, stream>>>(pooled, dec_W, dec_b, out);
}
